// AttentionLayer_24687472017803
// MI455X (gfx1250) — matrix operand order repaired and checked by reading, not compiled, not run on hardware
//
#include <hip/hip_runtime.h>

#define DEV __device__ __forceinline__

typedef unsigned short u16;
typedef __attribute__((ext_vector_type(16))) __bf16          v16bf;
typedef __attribute__((ext_vector_type(16))) unsigned short  v16u;
typedef __attribute__((ext_vector_type(8)))  float           v8f;
typedef __attribute__((ext_vector_type(4)))  unsigned int    v4u;
typedef __attribute__((ext_vector_type(4)))  float           v4f;

static constexpr int   Bb = 2, Nn = 1024, Dd = 1024, Hh = 16, Ee = 64;
static constexpr float SCALE = 0.125f;   // 1/sqrt(64)

union FragU { v4u q[2]; v16u u; v16bf b; };

DEV u16 f2bf(float x) {
    union { float f; unsigned u; } v; v.f = x;
    unsigned r = v.u + 0x7FFFu + ((v.u >> 16) & 1u);
    return (u16)(r >> 16);
}

// A-fragment 16x32 bf16 from row-major [16 x >=32] tile (base at tile origin).
// lane<16: row=lane, K in {0..7, 16..23}; lane>=16: row=lane-16, K in {8..15, 24..31}
DEV v16bf fragA(const u16* base, int ld, int lane) {
    int m  = lane & 15;
    int k0 = (lane & 16) ? 8 : 0;
    FragU f;
    f.q[0] = *(const v4u*)(base + m * ld + k0);
    f.q[1] = *(const v4u*)(base + m * ld + k0 + 16);
    return f.b;
}

// B-fragment 32x16 bf16 sourced from B^T stored row-major [16 x >=32]
// (row n = output column; lane<16: K in {0..7, 16..23}; lane>=16: K in {8..15, 24..31}).
DEV v16bf fragB(const u16* base, int ld, int lane) {
    int n  = lane & 15;
    int k0 = (lane & 16) ? 8 : 0;
    const v4u* p = (const v4u*)(base + n * ld + k0);
    FragU f;
    f.q[0] = p[0];
    f.q[1] = p[2];
    return f.b;
}

// A-fragment built from fp32 probabilities in LDS (quantize to bf16 on the fly)
DEV v16bf fragA_lds(const float* S, int ld, int lane, int kbase) {
    int m  = lane & 15;
    int k0 = kbase + ((lane & 16) ? 8 : 0);
    v16u u;
#pragma unroll
    for (int i = 0; i < 8; ++i) u[i]     = f2bf(S[m * ld + k0 + i]);
#pragma unroll
    for (int i = 0; i < 8; ++i) u[8 + i] = f2bf(S[m * ld + k0 + 16 + i]);
    FragU f; f.u = u;
    return f.b;
}

DEV v8f wmma_bf16(v16bf a, v16bf b, v8f c) {
    return __builtin_amdgcn_wmma_f32_16x16x32_bf16(false, a, false, b, (short)0, c, false, false);
}

// -------- CDNA5 async global->LDS copy (ASYNCcnt-tracked), inline asm --------
DEV void async_ld_b128(unsigned lds_addr, unsigned long long gaddr) {
    asm volatile("global_load_async_to_lds_b128 %0, %1, off"
                 :: "v"(lds_addr), "v"(gaddr) : "memory");
}
DEV void wait_async0() { asm volatile("s_wait_asynccnt 0x0" ::: "memory"); }
DEV void wait_async9() { asm volatile("s_wait_asynccnt 0x9" ::: "memory"); }

DEV unsigned lds_off(const void* p) { return (unsigned)(size_t)p; }

DEV float block_reduce_sum(float v, float* red, int tid, int nthr) {
    red[tid] = v; __syncthreads();
    for (int s = nthr >> 1; s > 0; s >>= 1) {
        if (tid < s) red[tid] += red[tid + s];
        __syncthreads();
    }
    float r = red[0]; __syncthreads();
    return r;
}

// Issue one double-buffer stage: A tile 32x32 (2KB) + B tile 256x32 (16KB).
// Each wave issues exactly 9 async b128 instructions (1 for A, 8 for B).
DEV void issue_stage(const u16* __restrict__ Ag, const u16* __restrict__ Wg,
                     int row0, int colblk, int k,
                     unsigned ldsA, unsigned ldsB, int w, int lane) {
    {
        unsigned fb  = (unsigned)(w * 512 + lane * 16);      // byte index in 2KB A tile
        unsigned row = fb >> 6, colb = fb & 63;              // 64B per row (32 bf16)
        const u16* g = Ag + (size_t)(row0 + row) * Dd + k + (colb >> 1);
        async_ld_b128(ldsA + fb, (unsigned long long)(size_t)g);
    }
#pragma unroll
    for (int i = 0; i < 8; ++i) {
        unsigned fb  = (unsigned)(w * 4096 + i * 512 + lane * 16); // byte idx in 16KB B tile
        unsigned row = fb >> 6, colb = fb & 63;
        const u16* g = Wg + (size_t)(colblk + row) * Dd + k + (colb >> 1);
        async_ld_b128(ldsB + fb, (unsigned long long)(size_t)g);
    }
}

// ---------------- Kernel 0: weight conversion / transposition to bf16 ----------------
__global__ void k_wconv(const float* __restrict__ q, const float* __restrict__ k,
                        const float* __restrict__ v, const float* __restrict__ g,
                        const float* __restrict__ ow,
                        u16* __restrict__ Wqkvg, u16* __restrict__ Wout) {
    int bx = blockIdx.x, tid = threadIdx.x;
    int d0 = tid * 4;
    if (bx < 4096) {
        int c = bx, p = c >> 10, h = (c >> 6) & 15, e = c & 63;
        const float* src = (p == 0) ? q : (p == 1) ? k : (p == 2) ? v : g;
#pragma unroll
        for (int i = 0; i < 4; ++i) {
            int d = d0 + i;
            Wqkvg[(size_t)c * Dd + d] = f2bf(src[((size_t)h * Dd + d) * Ee + e]);
        }
    } else {
        int c = bx - 4096;
#pragma unroll
        for (int i = 0; i < 4; ++i) {
            int kk = d0 + i;
            Wout[(size_t)c * Dd + kk] = f2bf(ow[(size_t)kk * Dd + c]);
        }
    }
}

// ---------------- Kernel 1: pre-LN of x -> bf16 activations ----------------
__global__ void k_ln_in(const float* __restrict__ x, const float* __restrict__ g,
                        const float* __restrict__ be, u16* __restrict__ Abf) {
    __shared__ float red[256];
    int row = blockIdx.x, tid = threadIdx.x;
    const float* xr = x + (size_t)row * Dd;
    v4f v = ((const v4f*)xr)[tid];
    float mu = block_reduce_sum(v[0] + v[1] + v[2] + v[3], red, tid, 256) * (1.0f / Dd);
    float sq = 0.f;
#pragma unroll
    for (int i = 0; i < 4; ++i) { float d = v[i] - mu; sq += d * d; }
    float var = block_reduce_sum(sq, red, tid, 256) * (1.0f / Dd);
    float rs = rsqrtf(var + 1e-6f);
    int d0 = tid * 4;
    u16* o = Abf + (size_t)row * Dd + d0;
#pragma unroll
    for (int i = 0; i < 4; ++i)
        o[i] = f2bf((v[i] - mu) * rs * g[d0 + i] + be[d0 + i]);
}

// ---------------- Kernel 2: fused Q/K/V/G projection GEMM ----------------
// 32x256 tile per WG; async double-buffered LDS staging; 8 WMMAs per wave per K-step.
__global__ void k_gemm_qkvg(const u16* __restrict__ A, const u16* __restrict__ W,
                            u16* __restrict__ Qb, u16* __restrict__ Kb,
                            u16* __restrict__ Vt, float* __restrict__ Gf) {
    __shared__ u16 shA[2][32 * 32];
    __shared__ u16 shB[2][256 * 32];
    int tid = threadIdx.x, lane = tid & 31, w = tid >> 5;
    int row0   = blockIdx.x * 32;
    int colblk = blockIdx.y * 256;
    int col0   = colblk + w * 64;
    unsigned baseA = lds_off(&shA[0][0]);
    unsigned baseB = lds_off(&shB[0][0]);

    v8f Z = {0.f, 0.f, 0.f, 0.f, 0.f, 0.f, 0.f, 0.f};
    v8f acc[2][4] = {{Z, Z, Z, Z}, {Z, Z, Z, Z}};

    issue_stage(A, W, row0, colblk, 0, baseA, baseB, w, lane);
    for (int kt = 0; kt < 32; ++kt) {
        int buf = kt & 1, nbuf = buf ^ 1;
        if (kt + 1 < 32) {
            issue_stage(A, W, row0, colblk, (kt + 1) * 32,
                        baseA + (unsigned)nbuf * (32 * 32 * 2),
                        baseB + (unsigned)nbuf * (256 * 32 * 2), w, lane);
            wait_async9();           // previous stage's 9 copies complete
        } else {
            wait_async0();
        }
        __syncthreads();
        const u16* sA = &shA[buf][0];
        const u16* sB = &shB[buf][0];
        v16bf a0 = fragA(sA,            32, lane);
        v16bf a1 = fragA(sA + 16 * 32,  32, lane);
#pragma unroll
        for (int j = 0; j < 4; ++j) {
            v16bf b = fragB(sB + (w * 64 + j * 16) * 32, 32, lane);
            acc[0][j] = wmma_bf16(a0, b, acc[0][j]);
            acc[1][j] = wmma_bf16(a1, b, acc[1][j]);
        }
        __syncthreads();
    }

    int n = lane & 15, m0 = (lane & 16) ? 8 : 0;
#pragma unroll
    for (int s = 0; s < 2; ++s) {
#pragma unroll
        for (int j = 0; j < 4; ++j) {
            int c = col0 + j * 16 + n;
            int p = c >> 10, h = (c >> 6) & 15, e = c & 63;
#pragma unroll
            for (int r = 0; r < 8; ++r) {
                int row = row0 + s * 16 + m0 + r;
                int bb = row >> 10, tok = row & 1023;
                float val = acc[s][j][r];
                size_t qidx = ((size_t)(bb * Hh + h) * Nn + tok) * Ee + e;
                if (p == 0)      Qb[qidx] = f2bf(val);
                else if (p == 1) Kb[qidx] = f2bf(val);
                else if (p == 2) Vt[((size_t)(bb * Hh + h) * Ee + e) * Nn + tok] = f2bf(val);
                else             Gf[qidx] = val;
            }
        }
    }
}

// ---------------- Kernel 3: attention (QK^T -> masked softmax -> PV -> gate) ----------------
// one workgroup per (b, h, 16-query tile); 4 waves, each owns 256 keys
__global__ void k_attn(const u16* __restrict__ Qb, const u16* __restrict__ Kb,
                       const u16* __restrict__ Vt, const float* __restrict__ Gf,
                       const float* __restrict__ mask, u16* __restrict__ Og) {
    extern __shared__ float lds[];
    float* S     = lds;                 // 16 x 1024 scores / probs
    float* Op    = S + 16 * 1024;       // 4 x 16 x 64 partial outputs
    float* rmax  = Op + 4 * 16 * 64;    // 16 x 8
    float* rsum  = rmax + 128;          // 16 x 8
    float* rstat = rsum + 128;          // 16 x 2

    int tid = threadIdx.x, lane = tid & 31, w = tid >> 5;
    int qt = blockIdx.x & 63;
    int hh = (blockIdx.x >> 6) & 15;
    int bb = blockIdx.x >> 10;
    int q0 = qt * 16;

    const u16*   Qh = Qb + (size_t)(bb * Hh + hh) * Nn * Ee;
    const u16*   Kh = Kb + (size_t)(bb * Hh + hh) * Nn * Ee;
    const u16*   Vh = Vt + (size_t)(bb * Hh + hh) * Ee * Nn;
    const float* Gh = Gf + (size_t)(bb * Hh + hh) * Nn * Ee;
    const float* mk = mask + (size_t)bb * Nn;

    v16bf qa0 = fragA(Qh + (size_t)q0 * Ee + 0,  Ee, lane);
    v16bf qa1 = fragA(Qh + (size_t)q0 * Ee + 32, Ee, lane);

    v8f Z = {0.f, 0.f, 0.f, 0.f, 0.f, 0.f, 0.f, 0.f};
    int key0 = w * 256;

    // --- scores ---
    for (int kt = 0; kt < 16; ++kt) {
        int kb = key0 + kt * 16;
        v16bf b0 = fragB(Kh + (size_t)kb * Ee + 0,  Ee, lane);
        v16bf b1 = fragB(Kh + (size_t)kb * Ee + 32, Ee, lane);
        v8f s = Z;
        s = wmma_bf16(qa0, b0, s);
        s = wmma_bf16(qa1, b1, s);
        int key = kb + (lane & 15);
        float madd = (mk[key] < -1.0f) ? -1e9f : 0.0f;   // key padding mask
        int m0 = (lane & 16) ? 8 : 0;
#pragma unroll
        for (int r = 0; r < 8; ++r)
            S[(m0 + r) * 1024 + key] = s[r] * SCALE + madd;
    }
    __syncthreads();

    // --- softmax over full 1024-key rows (8 threads per row) ---
    int row = tid & 15, seg = tid >> 4, cb = seg * 128;
    {
        float mx = -3.0e38f;
        for (int c = 0; c < 128; ++c) mx = fmaxf(mx, S[row * 1024 + cb + c]);
        rmax[row * 8 + seg] = mx;
    }
    __syncthreads();
    if (seg == 0) {
        float mx = rmax[row * 8];
        for (int i = 1; i < 8; ++i) mx = fmaxf(mx, rmax[row * 8 + i]);
        rstat[row * 2] = mx;
    }
    __syncthreads();
    {
        float mx = rstat[row * 2], ssum = 0.f;
        for (int c = 0; c < 128; ++c) {
            float e = __expf(S[row * 1024 + cb + c] - mx);
            S[row * 1024 + cb + c] = e;
            ssum += e;
        }
        rsum[row * 8 + seg] = ssum;
    }
    __syncthreads();
    if (seg == 0) {
        float ssum = 0.f;
        for (int i = 0; i < 8; ++i) ssum += rsum[row * 8 + i];
        rstat[row * 2 + 1] = 1.0f / ssum;
    }
    __syncthreads();
    {
        float inv = rstat[row * 2 + 1];
        for (int c = 0; c < 128; ++c) S[row * 1024 + cb + c] *= inv;
    }
    __syncthreads();

    // --- O += P @ V over this wave's 256 keys ---
    v8f acc[4] = {Z, Z, Z, Z};
    for (int kk = 0; kk < 8; ++kk) {
        int kbase = key0 + kk * 32;
        v16bf pa = fragA_lds(S, 1024, lane, kbase);
#pragma unroll
        for (int j = 0; j < 4; ++j) {
            v16bf vb = fragB(Vh + (size_t)(j * 16) * Nn + kbase, Nn, lane);
            acc[j] = wmma_bf16(pa, vb, acc[j]);
        }
    }
    {
        int n = lane & 15, m0 = (lane & 16) ? 8 : 0;
#pragma unroll
        for (int j = 0; j < 4; ++j)
#pragma unroll
            for (int r = 0; r < 8; ++r)
                Op[(w * 16 + m0 + r) * 64 + j * 16 + n] = acc[j][r];
    }
    __syncthreads();

    // --- reduce 4 wave partials, gate, query-mask, store bf16 in (b,n,h*E+e) ---
    for (int i = tid; i < 16 * 64; i += 128) {
        int m = i >> 6, e = i & 63;
        float o = Op[i] + Op[i + 1024] + Op[i + 2048] + Op[i + 3072];
        int tok = q0 + m;
        float gate = 1.0f / (1.0f + __expf(-Gh[(size_t)tok * Ee + e]));
        float qm = (mk[tok] < -1.0f) ? 0.0f : 1.0f;
        Og[((size_t)(bb * Nn + tok)) * Dd + hh * Ee + e] = f2bf(o * gate * qm);
    }
}

// ---------------- Kernel 4: output projection + bias + residual ----------------
__global__ void k_gemm_out(const u16* __restrict__ A, const u16* __restrict__ W,
                           const float* __restrict__ bias, const float* __restrict__ xres,
                           float* __restrict__ Y) {
    __shared__ u16 shA[2][32 * 32];
    __shared__ u16 shB[2][256 * 32];
    int tid = threadIdx.x, lane = tid & 31, w = tid >> 5;
    int row0   = blockIdx.x * 32;
    int colblk = blockIdx.y * 256;
    int col0   = colblk + w * 64;
    unsigned baseA = lds_off(&shA[0][0]);
    unsigned baseB = lds_off(&shB[0][0]);

    v8f Z = {0.f, 0.f, 0.f, 0.f, 0.f, 0.f, 0.f, 0.f};
    v8f acc[2][4] = {{Z, Z, Z, Z}, {Z, Z, Z, Z}};

    issue_stage(A, W, row0, colblk, 0, baseA, baseB, w, lane);
    for (int kt = 0; kt < 32; ++kt) {
        int buf = kt & 1, nbuf = buf ^ 1;
        if (kt + 1 < 32) {
            issue_stage(A, W, row0, colblk, (kt + 1) * 32,
                        baseA + (unsigned)nbuf * (32 * 32 * 2),
                        baseB + (unsigned)nbuf * (256 * 32 * 2), w, lane);
            wait_async9();
        } else {
            wait_async0();
        }
        __syncthreads();
        const u16* sA = &shA[buf][0];
        const u16* sB = &shB[buf][0];
        v16bf a0 = fragA(sA,           32, lane);
        v16bf a1 = fragA(sA + 16 * 32, 32, lane);
#pragma unroll
        for (int j = 0; j < 4; ++j) {
            v16bf b = fragB(sB + (w * 64 + j * 16) * 32, 32, lane);
            acc[0][j] = wmma_bf16(a0, b, acc[0][j]);
            acc[1][j] = wmma_bf16(a1, b, acc[1][j]);
        }
        __syncthreads();
    }

    int n = lane & 15, m0 = (lane & 16) ? 8 : 0;
#pragma unroll
    for (int s = 0; s < 2; ++s) {
#pragma unroll
        for (int j = 0; j < 4; ++j) {
            int c = col0 + j * 16 + n;
            float bc = bias[c];
#pragma unroll
            for (int r = 0; r < 8; ++r) {
                int row = row0 + s * 16 + m0 + r;
                Y[(size_t)row * Dd + c] = acc[s][j][r] + bc + xres[(size_t)row * Dd + c];
            }
        }
    }
}

// ---------------- Kernel 5: double LayerNorm + final mask ----------------
__global__ void k_ln_out(const float* __restrict__ Y, const float* __restrict__ mask,
                         const float* __restrict__ g1, const float* __restrict__ b1,
                         const float* __restrict__ g2, const float* __restrict__ b2,
                         float* __restrict__ out) {
    __shared__ float red[256];
    int row = blockIdx.x, tid = threadIdx.x;
    const float* yr = Y + (size_t)row * Dd;
    v4f v = ((const v4f*)yr)[tid];
    float mu = block_reduce_sum(v[0] + v[1] + v[2] + v[3], red, tid, 256) * (1.0f / Dd);
    float sq = 0.f;
#pragma unroll
    for (int i = 0; i < 4; ++i) { float d = v[i] - mu; sq += d * d; }
    float var = block_reduce_sum(sq, red, tid, 256) * (1.0f / Dd);
    float rs = rsqrtf(var + 1e-6f);
    int d0 = tid * 4;
    float t[4];
#pragma unroll
    for (int i = 0; i < 4; ++i)
        t[i] = (v[i] - mu) * rs * g1[d0 + i] + b1[d0 + i];

    float mu2 = block_reduce_sum(t[0] + t[1] + t[2] + t[3], red, tid, 256) * (1.0f / Dd);
    float sq2 = 0.f;
#pragma unroll
    for (int i = 0; i < 4; ++i) { float d = t[i] - mu2; sq2 += d * d; }
    float var2 = block_reduce_sum(sq2, red, tid, 256) * (1.0f / Dd);
    float rs2 = rsqrtf(var2 + 1e-6f);

    int bb = row >> 10, tok = row & 1023;
    float qm = (mask[(size_t)bb * Nn + tok] < -1.0f) ? 0.0f : 1.0f;
    v4f o;
#pragma unroll
    for (int i = 0; i < 4; ++i)
        o[i] = ((t[i] - mu2) * rs2 * g2[d0 + i] + b2[d0 + i]) * qm;
    ((v4f*)(out + (size_t)row * Dd))[tid] = o;
}

// ---------------- host launcher ----------------
extern "C" void kernel_launch(void* const* d_in, const int* in_sizes, int n_in,
                              void* d_out, int out_size, void* d_ws, size_t ws_size,
                              hipStream_t stream) {
    const float* x    = (const float*)d_in[0];
    const float* mask = (const float*)d_in[1];
    const float* qp   = (const float*)d_in[2];
    const float* kp   = (const float*)d_in[3];
    const float* vp   = (const float*)d_in[4];
    const float* gp   = (const float*)d_in[5];
    const float* ow   = (const float*)d_in[6];
    const float* ob   = (const float*)d_in[7];
    const float* lng  = (const float*)d_in[8];
    const float* lnb  = (const float*)d_in[9];
    const float* lrg  = (const float*)d_in[10];
    const float* lrb  = (const float*)d_in[11];
    const float* log_ = (const float*)d_in[12];
    const float* lob  = (const float*)d_in[13];
    float* out = (float*)d_out;

    char* ws = (char*)d_ws;
    size_t off = 0;
    auto alloc = [&](size_t bytes) -> void* {
        void* p = ws + off;
        off = (off + bytes + 255) & ~(size_t)255;
        return p;
    };
    u16*   Wqkvg = (u16*)  alloc((size_t)4096 * 1024 * 2);
    u16*   Wout  = (u16*)  alloc((size_t)1024 * 1024 * 2);
    u16*   Abf   = (u16*)  alloc((size_t)2048 * 1024 * 2);
    u16*   Qb    = (u16*)  alloc((size_t)Bb * Hh * Nn * Ee * 2);
    u16*   Kb    = (u16*)  alloc((size_t)Bb * Hh * Nn * Ee * 2);
    u16*   Vt    = (u16*)  alloc((size_t)Bb * Hh * Nn * Ee * 2);
    float* Gf    = (float*)alloc((size_t)Bb * Hh * Nn * Ee * 4);
    u16*   Og    = (u16*)  alloc((size_t)2048 * 1024 * 2);
    float* Y     = (float*)alloc((size_t)2048 * 1024 * 4);

    k_wconv<<<5120, 256, 0, stream>>>(qp, kp, vp, gp, ow, Wqkvg, Wout);
    k_ln_in<<<2048, 256, 0, stream>>>(x, lng, lnb, Abf);
    k_gemm_qkvg<<<dim3(64, 16), 128, 0, stream>>>(Abf, Wqkvg, Qb, Kb, Vt, Gf);
    size_t sh = (size_t)(16 * 1024 + 4 * 16 * 64 + 128 + 128 + 32) * sizeof(float);
    k_attn<<<2048, 128, sh, stream>>>(Qb, Kb, Vt, Gf, mask, Og);
    k_gemm_out<<<dim3(64, 4), 128, 0, stream>>>(Og, Wout, ob, x, Y);
    k_ln_out<<<2048, 256, 0, stream>>>(Y, mask, lrg, lrb, log_, lob, out);
}
